// CRFLoss_18614388261360
// MI455X (gfx1250) — matrix operand order repaired and checked by reading, not compiled, not run on hardware
//
#include <hip/hip_runtime.h>
#include <stdint.h>

// ---------------- problem constants ----------------
#define NB   4
#define NC   21
#define CP   32          // padded channel count (ch31 == ones plane for wsum)
#define HH   256
#define WW   256
#define PADB 12          // spatial zero border (>= gauss radius 9, >= bilat radius 6)
#define HP   (HH + 2*PADB)
#define WP   (WW + 2*PADB)
#define GR   9           // gaussian radius
#define BR   6           // bilateral radius

typedef _Float16 f16;
typedef __attribute__((address_space(3))) _Float16 lds_f16;
typedef __attribute__((ext_vector_type(16))) _Float16 v16h;
typedef __attribute__((ext_vector_type(8)))  _Float16 half8;
typedef __attribute__((ext_vector_type(2)))  _Float16 h2;
typedef __attribute__((ext_vector_type(8)))  float    v8f;

static constexpr size_t QSZ   = (size_t)NB * HP * CP * WP;   // halves per planar buffer
static constexpr size_t IMGSZ = (size_t)NB * HP * WP * 4;    // halves (RGBA fp16)
static constexpr size_t UHSZ  = (size_t)NB * NC * HH * WW;   // halves (negU fp16)
static constexpr size_t ZERO_BYTES = 3 * QSZ * 2 + IMGSZ * 2;   // region that must be zeroed

// 19-tap sigma=3 gaussian, exp(-d^2/18), compile-time constants
constexpr float GW[19] = {
  0.0111090f, 0.0285652f, 0.0657288f, 0.1353353f, 0.2493525f,
  0.4111123f, 0.6065307f, 0.8007374f, 0.9459595f, 1.0000000f,
  0.9459595f, 0.8007374f, 0.6065307f, 0.4111123f, 0.2493525f,
  0.1353353f, 0.0657288f, 0.0285652f, 0.0111090f };
constexpr float gsum() { float s = 0.f; for (int i = 0; i < 19; ++i) s += GW[i]; return s; }
constexpr float GSUM   = gsum();
constexpr float INVGS  = 1.0f / GSUM;
constexpr float W0     = 1.0f / (GSUM * GSUM);       // self-interaction weight (k[r]^2)
constexpr float INV1W0 = 1.0f / (1.0f - W0);

__device__ __forceinline__ float frcp(float x) { return __builtin_amdgcn_rcpf(x); }

__device__ __forceinline__ size_t qidx(int b, int yp, int c, int xp) {
  return (((size_t)b * HP + yp) * CP + c) * WP + xp;
}
__device__ __forceinline__ size_t qrow(int b, int yp, int c) {
  return (((size_t)b * HP + yp) * CP + c) * WP;
}
__device__ __forceinline__ size_t uidx(int b, int c, int y, int x) {
  return (((size_t)b * NC + c) * HH + y) * WW + x;
}

// ---------------- image quantization: floor(clip(img*255)) -> fp16 RGBA, pixel-major padded
__global__ void k_init_img(const float* __restrict__ images, f16* __restrict__ imgp) {
  int id = blockIdx.x * blockDim.x + threadIdx.x;
  if (id >= NB * HH * WW) return;
  int x = id & 255, y = (id >> 8) & 255, b = id >> 16;
  size_t base = (((size_t)b * HP + (y + PADB)) * WP + (x + PADB)) * 4;
  size_t src  = (((size_t)b * 3) * HH + y) * WW + x;
  #pragma unroll
  for (int k = 0; k < 3; ++k) {
    float v = images[src + (size_t)k * HH * WW];
    v = floorf(fminf(fmaxf(v * 255.0f, 0.0f), 255.0f));
    imgp[base + k] = (f16)v;
  }
}

// ---------------- init: negU = log(clip(softmax(logits),1e-5,1)) -> uh; Q0 = softmax(negU) -> qh
// also sets the ones plane (ch31) EVERYWHERE (borders included) for the wsum trick
__global__ void k_init_q(const float* __restrict__ logits, f16* __restrict__ qh,
                         f16* __restrict__ uh) {
  int id = blockIdx.x * blockDim.x + threadIdx.x;
  if (id >= NB * HP * WP) return;
  int b = id / (HP * WP);
  int r = id % (HP * WP);
  int yp = r / WP, xp = r % WP;
  qh[qidx(b, yp, 31, xp)] = (f16)1.0f;
  if (yp < PADB || yp >= PADB + HH || xp < PADB || xp >= PADB + WW) return;
  int y = yp - PADB, x = xp - PADB;
  size_t lb = (((size_t)b * NC) * HH + y) * WW + x;
  const size_t cs = (size_t)HH * WW;
  float l[NC], nl[NC];
  float m = -1e30f;
  #pragma unroll
  for (int c = 0; c < NC; ++c) { l[c] = logits[lb + c * cs]; m = fmaxf(m, l[c]); }
  float S = 0.0f;
  #pragma unroll
  for (int c = 0; c < NC; ++c) S += __expf(l[c] - m);
  float invS = frcp(S);
  float m2 = -1e30f;
  #pragma unroll
  for (int c = 0; c < NC; ++c) {
    float sm = __expf(l[c] - m) * invS;
    sm = fminf(fmaxf(sm, 1e-5f), 1.0f);
    nl[c] = __logf(sm);              // -U
    uh[uidx(b, c, y, x)] = (f16)nl[c];
    m2 = fmaxf(m2, nl[c]);
  }
  float S2 = 0.0f;
  #pragma unroll
  for (int c = 0; c < NC; ++c) S2 += __expf(nl[c] - m2);
  float inv = frcp(S2);
  #pragma unroll
  for (int c = 0; c < NC; ++c)
    qh[qidx(b, yp, c, xp)] = (f16)(__expf(nl[c] - m2) * inv);
}

// ---------------- separable 19-tap gaussian, vertical pass: qh -> tmp (constant-folded taps)
__global__ void k_gaussH(const f16* __restrict__ qh, f16* __restrict__ tmp) {
  int id = blockIdx.x * blockDim.x + threadIdx.x;
  if (id >= NB * NC * HH * WW) return;
  int x = id & 255, y = (id >> 8) & 255;
  int c = (id >> 16) % NC, b = (id >> 16) / NC;
  const size_t ystride = (size_t)CP * WP;
  const f16* col = qh + qidx(b, y + PADB - GR, c, x + PADB);
  float acc = 0.0f;
  #pragma unroll
  for (int t = 0; t < 2 * GR + 1; ++t)
    acc += GW[t] * (float)col[(size_t)t * ystride];
  tmp[qidx(b, y + PADB, c, x + PADB)] = (f16)(acc * INVGS);
}

// ---------------- horizontal pass + self-term removal: (blur - w0*Q)/(1-w0) -> mg
__global__ void k_gaussW(const f16* __restrict__ tmp, const f16* __restrict__ qh,
                         f16* __restrict__ mg) {
  int id = blockIdx.x * blockDim.x + threadIdx.x;
  if (id >= NB * NC * HH * WW) return;
  int x = id & 255, y = (id >> 8) & 255;
  int c = (id >> 16) % NC, b = (id >> 16) / NC;
  const f16* row = tmp + qidx(b, y + PADB, c, x + PADB - GR);
  float acc = 0.0f;
  #pragma unroll
  for (int t = 0; t < 2 * GR + 1; ++t)
    acc += GW[t] * (float)row[t];
  size_t ci = qidx(b, y + PADB, c, x + PADB);
  float q = (float)qh[ci];
  mg[ci] = (f16)((acc * INVGS - W0 * q) * INV1W0);
}

// ---------------- bilateral message via WMMA (attention-style banded matmul)
// one wave per 16-pixel row tile; image tile staged once via async-to-LDS;
// 13 dy rows; 2x v_wmma_f32_16x16x32_f16 per dy
__global__ __launch_bounds__(32) void k_bilat(const f16* __restrict__ qh,
                                              const f16* __restrict__ imgp,
                                              f16* __restrict__ outmb) {
  __shared__ alignas(16) f16 simg[13][28][4];    // 13 neighbor rows x 28 px RGBA
  __shared__ alignas(16) f16 Wt[17][32];         // banded weight tile (+ dummy row 16)
  const int L  = threadIdx.x;
  const int bid = blockIdx.x;
  const int x0 = (bid & 15) << 4;
  const int y  = (bid >> 4) & 255;
  const int b  = bid >> 12;
  const int p  = L & 15;
  const int hi = L >> 4;                          // half-wave select

  // zero the weight tile (band entries are rewritten every dy; dummy row absorbs dx=+7)
  {
    uint32_t* wz = (uint32_t*)&Wt[0][0];
    for (int i = L; i < 17 * 32 / 2; i += 32) wz[i] = 0u;
  }

  // ---- stage the 13x28 RGBA image tile into LDS with async copies (ASYNCcnt path)
  {
    const int ybase = y - BR + PADB;
    const int xbase = x0 - BR + PADB;
    for (int i = L; i < 13 * 28; i += 32) {
      int row = i / 28, col = i % 28;
      const f16* gp = imgp + (((size_t)b * HP + (ybase + row)) * WP + (xbase + col)) * 4;
      uint32_t la = (uint32_t)(uintptr_t)(lds_f16*)&simg[row][col][0];
      unsigned long long ga = (unsigned long long)(uintptr_t)gp;
      asm volatile("global_load_async_to_lds_b64 %0, %1, off"
                   :: "v"(la), "v"(ga) : "memory");
    }
    asm volatile("s_wait_asynccnt 0" ::: "memory");
  }
  __syncthreads();

  // own pixel color (dx=0, dy=0 slot of the staged tile)
  float pr, pg, pbv;
  {
    h2 c01 = *(const h2*)&simg[BR][p + BR][0];
    pr = (float)c01[0]; pg = (float)c01[1];
    pbv = (float)simg[BR][p + BR][2];
  }

  v8f acc0 = {};  // channels 0..15
  v8f acc1 = {};  // channels 16..31 (col 15 = wsum via ones plane)

  for (int dy = -BR; dy <= BR; ++dy) {
    const int ypn = y + dy + PADB;
    const int dyi = dy + BR;
    // ---- banded weights W[p][p+dxi], dxi in [0,12]; dxi==13 redirected to dummy row
    #pragma unroll
    for (int t = 0; t < 7; ++t) {
      int dxi = 2 * t + hi;                 // 0..13
      int col = p + dxi; if (col > 27) col = 27;
      h2 n01 = *(const h2*)&simg[dyi][col][0];
      float dr = pr  - (float)n01[0];
      float dg = pg  - (float)n01[1];
      float db = pbv - (float)simg[dyi][col][2];
      int dx = dxi - BR;
      float e = (float)(dy * dy + dx * dx) + dr * dr + dg * dg + db * db;
      f16 w = (f16)__expf(-e * (1.0f / 338.0f));     // sxy = srgb = 13
      int rowSel = (dxi < 13) ? p : 16;
      Wt[rowSel][p + dxi] = w;
    }
    __syncthreads();

    // ---- A operand (16x32 f16): lane<16 holds K 0..15 of row p; lane>=16 K 16..31
    v16h A;
    {
      const int k0a = hi ? 16 : 0;
      half8 a0 = *(const half8*)&Wt[p][k0a];
      half8 a1 = *(const half8*)&Wt[p][k0a + 8];
      #pragma unroll
      for (int i = 0; i < 8; ++i) { A[i] = a0[i]; A[8 + i] = a1[i]; }
    }
    __syncthreads();   // protect Wt before next iteration's stores

    // ---- B operands (32x16 f16): lane<16 holds K 0..15, lane>=16 K 16..31 of its channel column
    const int baseK = hi ? 16 : 0;
    const size_t xs = (size_t)(x0 + 6 + baseK);   // (x0 - 6 + baseK) + PADB, even -> 4B aligned
    const h2* pq0 = (const h2*)(qh + qrow(b, ypn, p)      + xs);
    const h2* pq1 = (const h2*)(qh + qrow(b, ypn, 16 + p) + xs);
    v16h B0, B1;
    #pragma unroll
    for (int i = 0; i < 8; ++i) {
      h2 u0 = pq0[i], u1 = pq1[i];
      B0[2 * i] = u0[0]; B0[2 * i + 1] = u0[1];
      B1[2 * i] = u1[0]; B1[2 * i + 1] = u1[1];
    }

    acc0 = __builtin_amdgcn_wmma_f32_16x16x32_f16(false, A, false, B0, (short)0, acc0, false, false);
    acc1 = __builtin_amdgcn_wmma_f32_16x16x32_f16(false, A, false, B1, (short)0, acc1, false, false);
  }

  // ---- epilogue: subtract center tap (w=1), normalize by (wsum + 1e-8), store mb
  const int ch0 = p, ch1 = 16 + p;
  #pragma unroll
  for (int r = 0; r < 8; ++r) {
    int prow = r + (hi << 3);                         // C/D layout: M = r + 8*hi, N = lane&15
    float wa  = __shfl(acc1[r], (hi << 4) | 15, 32);  // wsum accumulator (col 15)
    float inv = frcp((wa - 1.0f) + 1e-8f);
    size_t ci = qidx(b, y + PADB, ch0, x0 + prow + PADB);
    float qc = (float)qh[ci];
    outmb[ci] = (f16)((acc0[r] - qc) * inv);
    if (ch1 <= 20) {
      size_t ci1 = qidx(b, y + PADB, ch1, x0 + prow + PADB);
      float qc1 = (float)qh[ci1];
      outmb[ci1] = (f16)((acc1[r] - qc1) * inv);
    }
  }
}

// ---------------- mean-field update: Q = softmax(negU + 3*mg + 10*mb)
__global__ void k_update(const f16* __restrict__ uh, const f16* __restrict__ mg,
                         const f16* __restrict__ mb, f16* __restrict__ qh) {
  int id = blockIdx.x * blockDim.x + threadIdx.x;
  if (id >= NB * HH * WW) return;
  int x = id & 255, y = (id >> 8) & 255, b = id >> 16;
  float nl[NC];
  float m2 = -1e30f;
  #pragma unroll
  for (int c = 0; c < NC; ++c) {
    size_t ci = qidx(b, y + PADB, c, x + PADB);
    float u = (float)uh[uidx(b, c, y, x)];
    nl[c] = u + 3.0f * (float)mg[ci] + 10.0f * (float)mb[ci];
    m2 = fmaxf(m2, nl[c]);
  }
  float S2 = 0.0f;
  #pragma unroll
  for (int c = 0; c < NC; ++c) S2 += __expf(nl[c] - m2);
  float inv = frcp(S2);
  #pragma unroll
  for (int c = 0; c < NC; ++c)
    qh[qidx(b, y + PADB, c, x + PADB)] = (f16)(__expf(nl[c] - m2) * inv);
  qh[qidx(b, y + PADB, 31, x + PADB)] = (f16)1.0f;   // keep ones plane
}

// ---------------- loss: -mean log_softmax(log(Q+1e-8))[label]
__global__ void k_loss(const f16* __restrict__ qh, const int* __restrict__ labels,
                       float* __restrict__ partial) {
  __shared__ float smem[256];
  int tid = threadIdx.x;
  int id = blockIdx.x * 256 + tid;
  int x = id & 255, y = (id >> 8) & 255, b = id >> 16;
  int lab = labels[id];
  lab = lab < 0 ? 0 : (lab > 20 ? 20 : lab);
  float S = 0.0f, ql = 1e-8f;
  #pragma unroll
  for (int c = 0; c < NC; ++c) {
    float qv = (float)qh[qidx(b, y + PADB, c, x + PADB)] + 1e-8f;
    S += qv;
    if (c == lab) ql = qv;
  }
  smem[tid] = -(__logf(ql) - __logf(S));
  __syncthreads();
  for (int s = 128; s > 0; s >>= 1) {
    if (tid < s) smem[tid] += smem[tid + s];
    __syncthreads();
  }
  if (tid == 0) partial[blockIdx.x] = smem[0];
}

__global__ void k_final(const float* __restrict__ partial, float* __restrict__ out) {
  __shared__ float smem[256];
  int tid = threadIdx.x;
  float s = 0.0f;
  for (int i = tid; i < 1024; i += 256) s += partial[i];
  smem[tid] = s;
  __syncthreads();
  for (int k = 128; k > 0; k >>= 1) {
    if (tid < k) smem[tid] += smem[tid + k];
    __syncthreads();
  }
  if (tid == 0) out[0] = smem[0] * (1.0f / (float)(NB * HH * WW));
}

// ---------------- host launcher (graph-capture safe, deterministic) ----------------
extern "C" void kernel_launch(void* const* d_in, const int* in_sizes, int n_in,
                              void* d_out, int out_size, void* d_ws, size_t ws_size,
                              hipStream_t stream) {
  const float* logits = (const float*)d_in[0];
  const float* images = (const float*)d_in[1];
  const int*   labels = (const int*)d_in[2];
  float* out = (float*)d_out;

  f16* qh  = (f16*)d_ws;
  f16* tmp = qh  + QSZ;   // blurH scratch, then reused as mb
  f16* mg  = tmp + QSZ;
  f16* img = mg  + QSZ;
  f16* uh  = img + IMGSZ;
  float* partial = (float*)(uh + UHSZ);

  hipMemsetAsync(d_ws, 0, ZERO_BYTES, stream);   // zero padded Q/tmp/mg borders + image borders

  k_init_img<<<(NB * HH * WW + 255) / 256, 256, 0, stream>>>(images, img);
  k_init_q<<<(NB * HP * WP + 255) / 256, 256, 0, stream>>>(logits, qh, uh);

  const int nconv = NB * NC * HH * WW;
  for (int it = 0; it < 5; ++it) {
    k_gaussH<<<(nconv + 255) / 256, 256, 0, stream>>>(qh, tmp);
    k_gaussW<<<(nconv + 255) / 256, 256, 0, stream>>>(tmp, qh, mg);
    k_bilat<<<NB * HH * (WW / 16), 32, 0, stream>>>(qh, img, tmp);   // tmp := mb
    k_update<<<(NB * HH * WW + 255) / 256, 256, 0, stream>>>(uh, mg, tmp, qh);
  }

  k_loss<<<1024, 256, 0, stream>>>(qh, labels, partial);
  k_final<<<1, 256, 0, stream>>>(partial, out);
}
